// SpaMM_79310866088429
// MI455X (gfx1250) — matrix operand order repaired and checked by reading, not compiled, not run on hardware
//
#include <hip/hip_runtime.h>
#include <hip/hip_bf16.h>

typedef __attribute__((ext_vector_type(16))) _Float16 v16h;
typedef __attribute__((ext_vector_type(8)))  _Float16 v8h;
typedef __attribute__((ext_vector_type(8)))  float    v8f;
typedef __attribute__((ext_vector_type(4)))  unsigned int v4u;
typedef __attribute__((ext_vector_type(8)))  int      v8i;
typedef __attribute__((ext_vector_type(4)))  int      v4i;

#define NNODES 6000
#define CH 256
#define NEDGE 96000

#if __has_builtin(__builtin_amdgcn_tensor_load_to_lds)
#define HAVE_TDM 1
#endif

// ---------------------------------------------------------------------------
// WMMA fragment loaders (vectorized f32 -> f16 fragments), contiguous k = 16h + s order
// ---------------------------------------------------------------------------

// A fragment 16x32 (MxK): per lane two 8-float contiguous runs -> 4x b128
__device__ __forceinline__ v16h load_a_f32(const float* A, int lda, int m0, int k0) {
  const int lane = threadIdx.x & 31;
  const int m = m0 + (lane & 15);
  const int h = lane >> 4;
  const float* base = A + (size_t)m * lda + k0;
  const float4* p0 = (const float4*)(base + 16 * h);        // K = 16h .. 16h+7
  const float4* p1 = (const float4*)(base + 16 * h + 8);    // K = 16h+8 ..
  float4 x0 = p0[0], x1 = p0[1], y0 = p1[0], y1 = p1[1];
  v16h a;
  a[0] = (_Float16)x0.x;  a[1] = (_Float16)x0.y;  a[2] = (_Float16)x0.z;  a[3] = (_Float16)x0.w;
  a[4] = (_Float16)x1.x;  a[5] = (_Float16)x1.y;  a[6] = (_Float16)x1.z;  a[7] = (_Float16)x1.w;
  a[8] = (_Float16)y0.x;  a[9] = (_Float16)y0.y;  a[10] = (_Float16)y0.z; a[11] = (_Float16)y0.w;
  a[12] = (_Float16)y1.x; a[13] = (_Float16)y1.y; a[14] = (_Float16)y1.z; a[15] = (_Float16)y1.w;
  return a;
}

// B fragment 32x16 (KxN) with B[k][n] = S[n][k]: per lane 16 contiguous floats
__device__ __forceinline__ v16h load_bT_f32(const float* S, int ld, int n0, int k0) {
  const int lane = threadIdx.x & 31;
  const int n = n0 + (lane & 15);
  const int h = lane >> 4;
  const float4* p = (const float4*)(S + (size_t)n * ld + k0 + 16 * h);
  float4 x0 = p[0], x1 = p[1], x2 = p[2], x3 = p[3];
  v16h b;
  b[0] = (_Float16)x0.x;  b[1] = (_Float16)x0.y;  b[2] = (_Float16)x0.z;  b[3] = (_Float16)x0.w;
  b[4] = (_Float16)x1.x;  b[5] = (_Float16)x1.y;  b[6] = (_Float16)x1.z;  b[7] = (_Float16)x1.w;
  b[8] = (_Float16)x2.x;  b[9] = (_Float16)x2.y;  b[10] = (_Float16)x2.z; b[11] = (_Float16)x2.w;
  b[12] = (_Float16)x3.x; b[13] = (_Float16)x3.y; b[14] = (_Float16)x3.z; b[15] = (_Float16)x3.w;
  return b;
}

// ---------------------------------------------------------------------------
// Linear: C[M,Nc] (+)= A[M,256] @ W[:, kw0:kw0+256]^T  (+ bias, + activation)
// block = 128 threads (4 waves), grid = (M/16, Nc/64); M = 6000, K = 256
// ---------------------------------------------------------------------------
__global__ void gemm_wmma_kernel(const float* __restrict__ A, int lda,
                                 const float* __restrict__ W, int ldw, int kw0,
                                 const float* __restrict__ bias,
                                 float* __restrict__ C, int ldc,
                                 int accumulate, int act) {
  const int wave = threadIdx.x >> 5;
  const int lane = threadIdx.x & 31;
  const int m0 = blockIdx.x * 16;
  const int n0 = blockIdx.y * 64 + wave * 16;
  const int n = n0 + (lane & 15);
  const int h = lane >> 4;

  v8f acc = {};
  if (accumulate) {
#pragma unroll
    for (int r = 0; r < 8; ++r) acc[r] = C[(size_t)(m0 + r + 8 * h) * ldc + n];
  }
#pragma unroll
  for (int k0 = 0; k0 < 256; k0 += 32) {
    v16h a = load_a_f32(A, lda, m0, k0);
    v16h b = load_bT_f32(W, ldw, n0, kw0 + k0);
    acc = __builtin_amdgcn_wmma_f32_16x16x32_f16(false, a, false, b, (short)0,
                                                 acc, false, false);
  }
#pragma unroll
  for (int r = 0; r < 8; ++r) {
    float v = acc[r];
    if (bias) v += bias[n];
    if (act == 1) v = 1.0f / (1.0f + __expf(-v));
    else if (act == 2) v = tanhf(v);
    C[(size_t)(m0 + r + 8 * h) * ldc + n] = v;
  }
}

// ---------------------------------------------------------------------------
// Fused flash attention: O = softmax(Q @ Km^T * scale) @ V   (all [N,256])
// block = 128 (4 waves), grid = (N/16, 2); each wave: 16 q-rows x 32 out-cols
// K tile staged to LDS by the Tensor Data Mover; V tile staged transposed f16.
// ---------------------------------------------------------------------------
__global__ void flash_attn_kernel(const float* __restrict__ Q,
                                  const float* __restrict__ Km,
                                  const float* __restrict__ V,
                                  float* __restrict__ O,
                                  int Nk, float scale) {
  const int wave = threadIdx.x >> 5;
  const int lane = threadIdx.x & 31;
  const int m0 = blockIdx.x * 16;
  const int n0 = (blockIdx.y * 4 + wave) * 32;
  const int h = lane >> 4;
  const int nlow = lane & 15;

  __shared__ __align__(128) float    KtileF[32 * 256];   // 32 KB, f32 rows k0..k0+31
  __shared__ __align__(128) _Float16 VtileT[256 * 32];   // 16 KB, [col][key] f16
  __shared__ __align__(128) _Float16 Psm[4][16 * 32];    // 4 KB, P relayout per wave
  _Float16* sp = Psm[wave];

  v16h qf[8];
#pragma unroll
  for (int kc = 0; kc < 8; ++kc) qf[kc] = load_a_f32(Q, CH, m0, kc * 32);

  v8f acc0 = {}, acc1 = {};
  float mrun[8], lrun[8];
#pragma unroll
  for (int r = 0; r < 8; ++r) { mrun[r] = -3.0e38f; lrun[r] = 0.0f; }

  for (int k0 = 0; k0 < Nk; k0 += 32) {
    if (k0 + 32 < Nk) __builtin_prefetch(Km + (size_t)(k0 + 32) * CH, 0, 1);
    // ---- stage V tile transposed (all 128 threads) ----
    for (int i = threadIdx.x; i < 32 * 256; i += 128) {
      int r = i >> 8, c = i & 255;
      int gr = min(k0 + r, Nk - 1);
      VtileT[c * 32 + r] = (_Float16)V[(size_t)gr * CH + c];
    }
    // ---- stage K tile (f32) via Tensor Data Mover ----
#ifdef HAVE_TDM
    if (wave == 0) {
      unsigned long long ga =
          (unsigned long long)(uintptr_t)Km + (unsigned long long)k0 * CH * 4ull;
      unsigned int ldsOff = (unsigned int)(uintptr_t)(&KtileF[0]);
      int rows = Nk - k0;                       // OOB rows return zero
      v4u g0;
      g0[0] = 1u;                               // count=1, gather off
      g0[1] = ldsOff;                           // lds_addr (bytes)
      g0[2] = (unsigned int)ga;                 // global_addr[31:0]
      g0[3] = ((unsigned int)(ga >> 32) & 0x01FFFFFFu) | 0x80000000u; // addr[56:32] | type=2
      v8i g1;
      g1[0] = 0x00020000;                       // data_size=4B, wg_mask=0
      g1[1] = (int)(0x01000000u);               // tensor_dim0=256 (low16 -> bits 63:48)
      g1[2] = (int)(((unsigned)rows & 0xFFFFu) << 16); // tensor_dim1 low16
      g1[3] = (int)(0x01000000u);               // tile_dim0=256 (bits 127:112)
      g1[4] = 32;                               // tile_dim1=32
      g1[5] = 256;                              // tensor_dim0_stride (low 32)
      g1[6] = 0;
      g1[7] = 0;
      v4i gz; gz[0] = 0; gz[1] = 0; gz[2] = 0; gz[3] = 0;
      v8i gz8; gz8[0] = 0; gz8[1] = 0; gz8[2] = 0; gz8[3] = 0;
      gz8[4] = 0; gz8[5] = 0; gz8[6] = 0; gz8[7] = 0;
      __builtin_amdgcn_tensor_load_to_lds(g0, g1, gz, gz, gz8, 0);
      __builtin_amdgcn_s_wait_tensorcnt(0);
    }
#else
    for (int i = threadIdx.x; i < 32 * 256; i += 128) {
      int r = i >> 8, c = i & 255;
      int gr = min(k0 + r, Nk - 1);
      KtileF[i] = Km[(size_t)gr * CH + c];
    }
#endif
    __syncthreads();

    const bool t1valid = (k0 + 16) < Nk;
    v8f s0 = {}, s1 = {};
#pragma unroll
    for (int kc = 0; kc < 8; ++kc) {
      // B-frag for keys k0..k0+15 from LDS (16 contiguous f32 per lane)
      const float4* p = (const float4*)(KtileF + (size_t)nlow * 256 + kc * 32 + 16 * h);
      float4 x0 = p[0], x1 = p[1], x2 = p[2], x3 = p[3];
      v16h b0;
      b0[0] = (_Float16)x0.x;  b0[1] = (_Float16)x0.y;  b0[2] = (_Float16)x0.z;  b0[3] = (_Float16)x0.w;
      b0[4] = (_Float16)x1.x;  b0[5] = (_Float16)x1.y;  b0[6] = (_Float16)x1.z;  b0[7] = (_Float16)x1.w;
      b0[8] = (_Float16)x2.x;  b0[9] = (_Float16)x2.y;  b0[10] = (_Float16)x2.z; b0[11] = (_Float16)x2.w;
      b0[12] = (_Float16)x3.x; b0[13] = (_Float16)x3.y; b0[14] = (_Float16)x3.z; b0[15] = (_Float16)x3.w;
      s0 = __builtin_amdgcn_wmma_f32_16x16x32_f16(false, qf[kc], false, b0,
                                                  (short)0, s0, false, false);
    }
#pragma unroll
    for (int kc = 0; kc < 8; ++kc) {
      const float4* p = (const float4*)(KtileF + (size_t)(16 + nlow) * 256 + kc * 32 + 16 * h);
      float4 x0 = p[0], x1 = p[1], x2 = p[2], x3 = p[3];
      v16h b1;
      b1[0] = (_Float16)x0.x;  b1[1] = (_Float16)x0.y;  b1[2] = (_Float16)x0.z;  b1[3] = (_Float16)x0.w;
      b1[4] = (_Float16)x1.x;  b1[5] = (_Float16)x1.y;  b1[6] = (_Float16)x1.z;  b1[7] = (_Float16)x1.w;
      b1[8] = (_Float16)x2.x;  b1[9] = (_Float16)x2.y;  b1[10] = (_Float16)x2.z; b1[11] = (_Float16)x2.w;
      b1[12] = (_Float16)x3.x; b1[13] = (_Float16)x3.y; b1[14] = (_Float16)x3.z; b1[15] = (_Float16)x3.w;
      s1 = __builtin_amdgcn_wmma_f32_16x16x32_f16(false, qf[kc], false, b1,
                                                  (short)0, s1, false, false);
    }
    // online softmax: rows live in 16-lane halves -> shfl_xor reductions
    float mx[8];
#pragma unroll
    for (int r = 0; r < 8; ++r) {
      float a = s0[r] * scale;
      float b = t1valid ? s1[r] * scale : -3.0e38f;
      s0[r] = a; s1[r] = b;
      mx[r] = fmaxf(a, b);
    }
#pragma unroll
    for (int r = 0; r < 8; ++r) {
#pragma unroll
      for (int off = 8; off >= 1; off >>= 1)
        mx[r] = fmaxf(mx[r], __shfl_xor(mx[r], off, 32));
    }
#pragma unroll
    for (int r = 0; r < 8; ++r) {
      float mnew = fmaxf(mrun[r], mx[r]);
      float corr = __expf(mrun[r] - mnew);
      mrun[r] = mnew;
      float p0 = __expf(s0[r] - mnew);
      float p1 = __expf(s1[r] - mnew);
      s0[r] = p0; s1[r] = p1;
      acc0[r] *= corr; acc1[r] *= corr;
      float sum = p0 + p1;
#pragma unroll
      for (int off = 8; off >= 1; off >>= 1) sum += __shfl_xor(sum, off, 32);
      lrun[r] = lrun[r] * corr + sum;
    }
    // P: C-fragment -> LDS -> A-fragment relayout (wave-local; s_wait_dscnt)
#pragma unroll
    for (int r = 0; r < 8; ++r) {
      int row = r + 8 * h;
      sp[row * 32 + nlow]      = (_Float16)s0[r];
      sp[row * 32 + 16 + nlow] = (_Float16)s1[r];
    }
    asm volatile("s_wait_dscnt 0" ::: "memory");
    v8h plo = *(const v8h*)&sp[nlow * 32 + 16 * h];
    v8h phi = *(const v8h*)&sp[nlow * 32 + 16 * h + 8];
    v16h pf = __builtin_shufflevector(plo, phi, 0, 1, 2, 3, 4, 5, 6, 7,
                                      8, 9, 10, 11, 12, 13, 14, 15);
    // V B-fragments: 16 contiguous f16 per lane from transposed tile
    v16h vb0 = *(const v16h*)&VtileT[(size_t)(n0 + nlow) * 32 + 16 * h];
    v16h vb1 = *(const v16h*)&VtileT[(size_t)(n0 + 16 + nlow) * 32 + 16 * h];
    acc0 = __builtin_amdgcn_wmma_f32_16x16x32_f16(false, pf, false, vb0, (short)0,
                                                  acc0, false, false);
    acc1 = __builtin_amdgcn_wmma_f32_16x16x32_f16(false, pf, false, vb1, (short)0,
                                                  acc1, false, false);
    __syncthreads();   // protect tiles before next stage
  }
#pragma unroll
  for (int r = 0; r < 8; ++r) {
    float inv = 1.0f / lrun[r];
    int m = m0 + r + 8 * h;
    O[(size_t)m * CH + n0 + nlow]      = acc0[r] * inv;
    O[(size_t)m * CH + n0 + 16 + nlow] = acc1[r] * inv;
  }
}

// ---------------------------------------------------------------------------
// GAT support kernels
// ---------------------------------------------------------------------------
__device__ __forceinline__ void atomicMaxFloat(float* addr, float val) {
  if (val >= 0.0f) atomicMax((int*)addr, __float_as_int(val));
  else             atomicMin((unsigned int*)addr, (unsigned int)__float_as_int(val));
}

__global__ void fill_kernel(float* p, float v, int n) {
  int i = blockIdx.x * blockDim.x + threadIdx.x;
  if (i < n) p[i] = v;
}

__global__ void rows_bias_kernel(float* out, const float* __restrict__ b, int C) {
  out[(size_t)blockIdx.x * C + threadIdx.x] = b[threadIdx.x];
}

// per-row dot product (wave32 per row): out[i*ostride+ooff] = act(X[i,:].w + b)
__global__ void rowdot_kernel(const float* __restrict__ X, int C,
                              const float* __restrict__ w, const float* __restrict__ b,
                              float* __restrict__ out, int ostride, int ooff, int act) {
  const int i = blockIdx.x, lane = threadIdx.x;
  float s = 0.0f;
  for (int c = lane; c < C; c += 32) s += X[(size_t)i * C + c] * w[c];
#pragma unroll
  for (int off = 16; off >= 1; off >>= 1) s += __shfl_xor(s, off, 32);
  if (lane == 0) {
    if (b) s += b[0];
    if (act == 1) s = 1.0f / (1.0f + __expf(-s));
    out[(size_t)i * ostride + ooff] = s;
  }
}

__global__ void gat_edge1(const int* __restrict__ src, const int* __restrict__ dst,
                          int E, int EE, const float* __restrict__ ss,
                          const float* __restrict__ sd, float* __restrict__ aedge,
                          float* __restrict__ mbuf) {
  int e = blockIdx.x * blockDim.x + threadIdx.x;
  if (e >= EE) return;
  int s, d;
  if (e < E) { s = src[e]; d = dst[e]; } else { s = d = e - E; }
  float a = ss[s] + sd[d];
  a = (a > 0.0f) ? a : 0.2f * a;   // leaky_relu 0.2
  aedge[e] = a;
  atomicMaxFloat(&mbuf[d], a);
}

__global__ void gat_edge2(const int* __restrict__ src, const int* __restrict__ dst,
                          int E, int EE, float* __restrict__ aedge,
                          const float* __restrict__ mbuf, float* __restrict__ zbuf) {
  int e = blockIdx.x * blockDim.x + threadIdx.x;
  if (e >= EE) return;
  int d = (e < E) ? dst[e] : (e - E);
  float ev = __expf(aedge[e] - mbuf[d]);
  aedge[e] = ev;
  atomicAdd(&zbuf[d], ev);
  (void)src;
}

__global__ void gat_edge3(const int* __restrict__ src, const int* __restrict__ dst,
                          int E, int EE, const float* __restrict__ aedge,
                          const float* __restrict__ zbuf, const float* __restrict__ xp,
                          float* __restrict__ out, int C) {
  int e = blockIdx.x;
  int s, d;
  if (e < E) { s = src[e]; d = dst[e]; } else { s = d = e - E; }
  float alpha = aedge[e] / (zbuf[d] + 1e-16f);
  int c = threadIdx.x;
  atomicAdd(&out[(size_t)d * C + c], alpha * xp[(size_t)s * C + c]);
  (void)EE;
}

// ---------------------------------------------------------------------------
// Elementwise / normalization kernels
// ---------------------------------------------------------------------------
__global__ void attn_combine_kernel(const float* __restrict__ a1, const float* __restrict__ a2,
                                    const float* __restrict__ c1, const float* __restrict__ c2,
                                    float* __restrict__ out, long n) {
  long idx = (long)blockIdx.x * blockDim.x + threadIdx.x;
  if (idx >= n) return;
  float s1 = c1[idx], s2 = c2[idx];             // already sigmoided
  float mx = fmaxf(s1, s2);
  float e1 = __expf(s1 - mx), e2 = __expf(s2 - mx);
  float inv = 1.0f / (e1 + e2);
  out[idx] = (e1 * a1[idx] + e2 * a2[idx]) * inv;
}

__global__ void sgu_kernel(const float* __restrict__ xin, const float* __restrict__ xu,
                           const float* __restrict__ g, const float* __restrict__ alpha,
                           float* __restrict__ out, long n) {
  long idx = (long)blockIdx.x * blockDim.x + threadIdx.x;
  if (idx >= n) return;
  out[idx] = xin[idx] + alpha[0] * g[idx] * xu[idx];
}

__global__ void layernorm_kernel(const float* __restrict__ X, const float* __restrict__ g,
                                 const float* __restrict__ b, float* __restrict__ out) {
  __shared__ float red[8];
  __shared__ float stats[2];
  const int i = blockIdx.x, t = threadIdx.x;   // block = 256
  float x = X[(size_t)i * CH + t];
  float s = x;
#pragma unroll
  for (int off = 16; off >= 1; off >>= 1) s += __shfl_xor(s, off, 32);
  if ((t & 31) == 0) red[t >> 5] = s;
  __syncthreads();
  if (t == 0) {
    float tot = 0;
    for (int j = 0; j < 8; ++j) tot += red[j];
    stats[0] = tot * (1.0f / CH);
  }
  __syncthreads();
  float d = x - stats[0];
  float s2 = d * d;
#pragma unroll
  for (int off = 16; off >= 1; off >>= 1) s2 += __shfl_xor(s2, off, 32);
  if ((t & 31) == 0) red[t >> 5] = s2;
  __syncthreads();
  if (t == 0) {
    float tot = 0;
    for (int j = 0; j < 8; ++j) tot += red[j];
    stats[1] = tot * (1.0f / CH);
  }
  __syncthreads();
  out[(size_t)i * CH + t] = d * rsqrtf(stats[1] + 1e-5f) * g[t] + b[t];
}

__global__ void msf_combine_kernel(const float* __restrict__ P0, const float* __restrict__ P1,
                                   const float* __restrict__ P2, const float* __restrict__ wl,
                                   float* __restrict__ out) {
  const int i = blockIdx.x, t = threadIdx.x;   // block = 256
  float w0 = wl[i * 3], w1 = wl[i * 3 + 1], w2 = wl[i * 3 + 2];
  float mx = fmaxf(w0, fmaxf(w1, w2));
  float e0 = __expf(w0 - mx), e1 = __expf(w1 - mx), e2 = __expf(w2 - mx);
  float inv = 1.0f / (e0 + e1 + e2);
  size_t idx = (size_t)i * CH + t;
  out[idx] = (e0 * P0[idx] + e1 * P1[idx] + e2 * P2[idx]) * inv;
}

__global__ void concat2_kernel(const float* __restrict__ r0, const float* __restrict__ r1,
                               float* __restrict__ cat) {
  const int i = blockIdx.x, t = threadIdx.x;   // block = 256
  cat[(size_t)i * 256 + t] = (t < 128) ? r0[(size_t)i * 128 + t]
                                       : r1[(size_t)i * 128 + (t - 128)];
}

// ---------------------------------------------------------------------------
// Host orchestration
// ---------------------------------------------------------------------------
extern "C" void kernel_launch(void* const* d_in, const int* in_sizes, int n_in,
                              void* d_out_v, int out_size, void* d_ws, size_t ws_size,
                              hipStream_t stream) {
  (void)in_sizes; (void)n_in; (void)out_size; (void)ws_size;
  const int N = NNODES, C = CH, E = NEDGE, EE = E + N;

  const float* omics = (const float*)d_in[0];
  const int* ei_sp  = (const int*)d_in[1];
  const int* ei_om1 = (const int*)d_in[2];
  const int* ei_om2 = (const int*)d_in[3];
  auto F = [&](int i) { return (const float*)d_in[i]; };

  float* out = (float*)d_out_v;
  const size_t NCf = (size_t)N * C;   // 1,536,000 floats
  const size_t o_rec0 = NCf;
  const size_t o_rec1 = NCf + (size_t)N * 128;
  const size_t o_embP = 2 * NCf;
  const size_t o_se   = 3 * NCf;      // scale_embs  (3 x NC)
  const size_t o_seP  = 6 * NCf;      // scale_embs_ (3 x NC)

  float* ws = (float*)d_ws;
  auto buf = [&](int i) { return ws + (size_t)i * NCf; };
  float* B_om1e = buf(0);  float* B_om2e = buf(1);
  float* Emb0 = buf(2);    float* Emb1 = buf(3);    float* Emb2 = buf(4);
  float* Spg  = buf(5);
  float* Qb = buf(6);  float* K1b = buf(7);  float* K2b = buf(8);  float* Vb = buf(9);
  float* Aab = buf(10); float* Abb = buf(11); float* C1b = buf(12); float* C2b = buf(13);
  float* Xp = buf(14);
  float* P0b = buf(15); float* P1b = buf(16); float* P2b = buf(17);
  float* Catb = buf(18);
  float* small = buf(19);
  float* ssrc = small;          float* sdst = ssrc + N;
  float* mbuf = sdst + N;       float* zbuf = mbuf + N;
  float* aedge = zbuf + N;      float* wl = aedge + (EE + 64);

  auto gemm = [&](const float* A, int lda, const float* W, int ldw, int kw0,
                  const float* bias, float* Cm, int Nc, int acc, int act) {
    dim3 g(N / 16, Nc / 64);
    gemm_wmma_kernel<<<g, 128, 0, stream>>>(A, lda, W, ldw, kw0, bias, Cm, Nc, acc, act);
  };

  // GAT: out = segment_softmax-weighted scatter of (X@W^T) + b
  auto gat = [&](const float* X, const int* ei, const float* Wg, const float* as_,
                 const float* ad_, const float* bg, float* outg, int Cout) {
    gemm(X, C, Wg, C, 0, nullptr, Xp, Cout, 0, 0);
    rowdot_kernel<<<N, 32, 0, stream>>>(Xp, Cout, as_, nullptr, ssrc, 1, 0, 0);
    rowdot_kernel<<<N, 32, 0, stream>>>(Xp, Cout, ad_, nullptr, sdst, 1, 0, 0);
    fill_kernel<<<(N + 255) / 256, 256, 0, stream>>>(mbuf, -3.0e38f, N);
    fill_kernel<<<(N + 255) / 256, 256, 0, stream>>>(zbuf, 0.0f, N);
    rows_bias_kernel<<<N, Cout, 0, stream>>>(outg, bg, Cout);
    const int* s_ = ei; const int* d_ = ei + E;
    gat_edge1<<<(EE + 255) / 256, 256, 0, stream>>>(s_, d_, E, EE, ssrc, sdst, aedge, mbuf);
    gat_edge2<<<(EE + 255) / 256, 256, 0, stream>>>(s_, d_, E, EE, aedge, mbuf, zbuf);
    gat_edge3<<<EE, Cout, 0, stream>>>(s_, d_, E, EE, aedge, zbuf, Xp, outg, Cout);
  };

  // sc_attn; mb = param base of mf[scale] (attn leaves at mb+0..mb+9)
  auto sc_attn = [&](int mb, const float* spg, float* outbuf) {
    gemm(spg,    C, F(mb + 7), C,   0,   F(mb + 6), Qb,  C, 0, 0);   // q
    gemm(B_om1e, C, F(mb + 3), C,   0,   F(mb + 2), K1b, C, 0, 0);   // k1
    gemm(B_om2e, C, F(mb + 5), C,   0,   F(mb + 4), K2b, C, 0, 0);   // k2
    // v = concat(sp, om1, om2) @ Wv^T + bv, as 3 K-chunk accumulations
    gemm(spg,    C, F(mb + 9), 768, 0,   nullptr,   Vb,  C, 0, 0);
    gemm(B_om1e, C, F(mb + 9), 768, 256, nullptr,   Vb,  C, 1, 0);
    gemm(B_om2e, C, F(mb + 9), 768, 512, F(mb + 8), Vb,  C, 1, 0);
    dim3 fg(N / 16, 2);
    flash_attn_kernel<<<fg, 128, 0, stream>>>(Qb, K1b, Vb, Aab, N, 0.0625f);
    flash_attn_kernel<<<fg, 128, 0, stream>>>(Qb, K2b, Vb, Abb, N, 0.0625f);
    // conf (both branches use conf1 — faithful to the source bug)
    gemm(spg,    C, F(mb + 1), 512, 0,   nullptr,   C1b, C, 0, 0);
    gemm(B_om1e, C, F(mb + 1), 512, 256, F(mb + 0), C1b, C, 1, 1);
    gemm(spg,    C, F(mb + 1), 512, 0,   nullptr,   C2b, C, 0, 0);
    gemm(B_om2e, C, F(mb + 1), 512, 256, F(mb + 0), C2b, C, 1, 1);
    attn_combine_kernel<<<(int)((NCf + 255) / 256), 256, 0, stream>>>(
        Aab, Abb, C1b, C2b, outbuf, (long)NCf);
  };

  // sgu; sb = param base of sgu[scale] (alpha, up_g.b, up_g.w, up_proj.b, up_proj.w)
  auto sgu = [&](int sb, const float* xin, const float* xup, float* outp) {
    float* xu = Qb; float* gb = K1b;
    gemm(xup, C, F(sb + 4), C,   0,   F(sb + 3), xu, C, 0, 2);       // tanh
    gemm(xin, C, F(sb + 2), 512, 0,   nullptr,   gb, C, 0, 0);
    gemm(xu,  C, F(sb + 2), 512, 256, F(sb + 1), gb, C, 1, 1);       // sigmoid
    sgu_kernel<<<(int)((NCf + 255) / 256), 256, 0, stream>>>(
        xin, xu, gb, F(sb + 0), outp, (long)NCf);
  };

  auto encoder = [&](const float* xin, float* emb_out, float* up0, float* up1, float* up2) {
    // mf[s] base = 12 + 14*s; gat leaves: +10 ad, +11 as, +12 b, +13 w
    gat(xin, ei_om1, F(12 + 13), F(12 + 11), F(12 + 10), F(12 + 12), B_om1e, C);
    gat(xin, ei_om2, F(12 + 13), F(12 + 11), F(12 + 10), F(12 + 12), B_om2e, C);
    gat(xin, ei_sp,  F(12 + 13), F(12 + 11), F(12 + 10), F(12 + 12), Spg, C);
    sc_attn(12, Spg, Emb0);
    gat(Emb0, ei_sp, F(26 + 13), F(26 + 11), F(26 + 10), F(26 + 12), Spg, C);
    sc_attn(26, Spg, Emb1);
    gat(Emb1, ei_sp, F(40 + 13), F(40 + 11), F(40 + 10), F(40 + 12), Spg, C);
    sc_attn(40, Spg, Emb2);
    sgu(72, Emb0, Emb0, up0);
    sgu(77, Emb1, Emb0, up1);
    sgu(82, Emb2, Emb1, up2);
    const float* ups[3] = {up0, up1, up2};
    float* Pr[3] = {P0b, P1b, P2b};
    for (int s = 0; s < 3; ++s) {   // msf[s] base = 54 + 6*s
      int mb = 54 + 6 * s;          // +0 ln_b, +1 ln_g, +2 proj.b, +3 proj.w, +4 wlin.b, +5 wlin.w
      gemm(ups[s], C, F(mb + 3), C, 0, F(mb + 2), Xp, C, 0, 0);
      layernorm_kernel<<<N, C, 0, stream>>>(Xp, F(mb + 1), F(mb + 0), Pr[s]);
      rowdot_kernel<<<N, 32, 0, stream>>>(Pr[s], C, F(mb + 5), F(mb + 4), wl, 3, s, 1);
    }
    msf_combine_kernel<<<N, C, 0, stream>>>(P0b, P1b, P2b, wl, emb_out);
  };

  // ---- full model ----
  encoder(omics, out, out + o_se, out + o_se + NCf, out + o_se + 2 * NCf);
  // decoder GATs (dec[0] leaves 4..7, dec[1] 8..11: ad, as, b, w), Cout=128
  gat(out, ei_sp, F(7),  F(5), F(4), F(6),  out + o_rec0, 128);
  gat(out, ei_sp, F(11), F(9), F(8), F(10), out + o_rec1, 128);
  concat2_kernel<<<N, 256, 0, stream>>>(out + o_rec0, out + o_rec1, Catb);
  encoder(Catb, out + o_embP, out + o_seP, out + o_seP + NCf, out + o_seP + 2 * NCf);
}
